// MultiHeadEncDec_40870908788990
// MI455X (gfx1250) — matrix operand order repaired and checked by reading, not compiled, not run on hardware
//
#include <hip/hip_runtime.h>
#include <hip/hip_bf16.h>

// MI455X / gfx1250. Flash-attention (softmax over the query index i, per the
// reference's axis=1 softmax) + projection, bf16 WMMA with f32 accumulation.
// One-time bf16 conversion/transpose prepass so the hot loops stage raw 16B
// chunks via the CDNA5 async global->LDS path (ASYNCcnt), double-buffered.

#define DM     128
#define SEQ    2048
#define NHEADS 64     // B * H
#define BATCH  4
#define HK     2048   // H * DM, projection K-dim

typedef __attribute__((ext_vector_type(16))) __bf16 v16bf;
typedef __attribute__((ext_vector_type(8)))  __bf16 v8bf;
typedef __attribute__((ext_vector_type(4)))  __bf16 v4bf;
typedef __attribute__((ext_vector_type(8)))  float  v8f;
typedef __attribute__((ext_vector_type(4)))  int    v4i;

#define AS1 __attribute__((address_space(1)))
#define AS3 __attribute__((address_space(3)))

#if defined(__has_builtin)
#if __has_builtin(__builtin_amdgcn_global_load_async_to_lds_b128) && \
    __has_builtin(__builtin_amdgcn_s_wait_asynccnt)
#define HAVE_ASYNC 1
#else
#define HAVE_ASYNC 0
#endif
#else
#define HAVE_ASYNC 0
#endif

__device__ inline void cp16(const __bf16* g, __bf16* l) {
#if HAVE_ASYNC
  __builtin_amdgcn_global_load_async_to_lds_b128((AS1 v4i*)g, (AS3 v4i*)l, 0, 0);
#else
  *(v8bf*)l = *(const v8bf*)g;
#endif
}

__device__ inline void wait_async() {
#if HAVE_ASYNC
  __builtin_amdgcn_s_wait_asynccnt(0);
#endif
}

__device__ inline v16bf join16(v8bf lo, v8bf hi) {
  v16bf r;
#pragma unroll
  for (int t = 0; t < 8; ++t) { r[t] = lo[t]; r[t + 8] = hi[t]; }
  return r;
}

// A/B fragment gather: 8+8 bf16 from one LDS row (both 16B-aligned).
__device__ inline v16bf ldfrag(const __bf16* row, int o_lo, int o_hi) {
  return join16(*(const v8bf*)&row[o_lo], *(const v8bf*)&row[o_hi]);
}

// ---------------------------------------------------------------------------
// Prepass 1: [N][DM][SEQ] f32  ->  [N][SEQ][DM] bf16 (transpose via LDS tile)
// ---------------------------------------------------------------------------
__global__ __launch_bounds__(256)
void cvt_t_kernel(const float* __restrict__ src, __bf16* __restrict__ dst) {
  __shared__ __align__(16) __bf16 st[64][136];
  const int n  = blockIdx.y;
  const int i0 = blockIdx.x * 64;
  const float* S = src + (size_t)n * DM * SEQ;
  __bf16* D = dst + (size_t)n * SEQ * DM;
  for (int e = threadIdx.x; e < 64 * DM; e += 256) {
    int d = e >> 6, i = e & 63;
    st[i][d] = (__bf16)S[(size_t)d * SEQ + i0 + i];
  }
  __syncthreads();
  for (int e = threadIdx.x; e < 64 * DM; e += 256) {
    int i = e >> 7, d = e & 127;
    D[(size_t)(i0 + i) * DM + d] = st[i][d];
  }
}

// Prepass 2: elementwise f32 -> bf16 (for V), 4 elems/thread
__global__ __launch_bounds__(256)
void cvt_v_kernel(const float* __restrict__ s, __bf16* __restrict__ d) {
  size_t i = ((size_t)blockIdx.x * 256 + threadIdx.x) * 4;
  float4 v = *(const float4*)(s + i);
  v4bf o;
  o[0] = (__bf16)v.x; o[1] = (__bf16)v.y; o[2] = (__bf16)v.z; o[3] = (__bf16)v.w;
  *(v4bf*)(d + i) = o;
}

// Prepass 3: W [HK][DM] f32 -> Wt [DM][HK] bf16
__global__ __launch_bounds__(256)
void cvt_w_kernel(const float* __restrict__ W, __bf16* __restrict__ Wt) {
  __shared__ __align__(16) __bf16 st[128][136];
  const int k0 = blockIdx.x * 128;
  for (int e = threadIdx.x; e < 128 * 128; e += 256) {
    int k = e >> 7, d = e & 127;
    st[k][d] = (__bf16)W[(size_t)(k0 + k) * DM + d];
  }
  __syncthreads();
  for (int e = threadIdx.x; e < 128 * 128; e += 256) {
    int d = e >> 7, k = e & 127;
    Wt[(size_t)d * HK + k0 + k] = st[k][d];
  }
}

// ---------------------------------------------------------------------------
// Kernel 1: fused scores + online softmax(axis=i) + V*P, per (head, j-block).
// Qt/Kt: [N][SEQ][DM] bf16, Vb: [N][DM][SEQ] bf16.
// OT out: [BATCH][SEQ(j)][HK(k=h*128+d)] bf16 (transposed for the projection).
// ---------------------------------------------------------------------------
__global__ __launch_bounds__(256)
void attn_kernel(const __bf16* __restrict__ Qt,
                 const __bf16* __restrict__ Kt,
                 const __bf16* __restrict__ Vb,
                 __bf16* __restrict__ OT)
{
  __shared__ __align__(16) __bf16 sK[128][136];    // [j_local][d]
  __shared__ __align__(16) __bf16 sQ[2][32][136];  // [i_local][d], double buffered
  __shared__ __align__(16) __bf16 sV[2][128][32];  // [d][i_local], double buffered

  const int n    = blockIdx.y;
  const int jb   = blockIdx.x * 128;
  const int tid  = threadIdx.x;
  const int w    = tid >> 5;
  const int lane = tid & 31;
  const int lj   = lane & 15;
  const int half = lane >> 4;

  const float scale = 0.08838834764831844f;   // 1/sqrt(128)

  const __bf16* Qtn = Qt + (size_t)n * SEQ * DM;
  const __bf16* Ktn = Kt + (size_t)n * SEQ * DM;
  const __bf16* Vbn = Vb + (size_t)n * DM * SEQ;

  // ---- stage K tile: 128 rows x 256B, pure 16B copies ----
#pragma unroll
  for (int t = 0; t < 8; ++t) {
    int e = tid + t * 256;
    int j = e >> 4, c = e & 15;
    cp16(&Ktn[(size_t)(jb + j) * DM + c * 8], &sK[j][c * 8]);
  }
  // ---- stage first Q/V chunk into buffer 0 ----
  {
#pragma unroll
    for (int t = 0; t < 2; ++t) {
      int e = tid + t * 256;
      int i = e >> 4, c = e & 15;
      cp16(&Qtn[(size_t)i * DM + c * 8], &sQ[0][i][c * 8]);
    }
#pragma unroll
    for (int t = 0; t < 2; ++t) {
      int e = tid + t * 256;
      int d = e >> 2, c = e & 3;
      cp16(&Vbn[(size_t)d * SEQ + c * 8], &sV[0][d][c * 8]);
    }
  }
  wait_async();
  __syncthreads();

  // Loop-invariant B-fragments of K (B[K=d, N=j]); 4 chunks of d=32.
  v16bf kf[4];
  const int jrow = w * 16 + lj;
#pragma unroll
  for (int c = 0; c < 4; ++c)
    kf[c] = ldfrag(&sK[jrow][0], c * 32 + half * 8, c * 32 + half * 8 + 16);

  v8f acc[8];
#pragma unroll
  for (int t = 0; t < 8; ++t)
#pragma unroll
    for (int r = 0; r < 8; ++r) acc[t][r] = 0.f;

  float m_run = -3.0e38f, l_run = 0.f;

  for (int it = 0; it < SEQ / 32; ++it) {
    const int cur = it & 1, nxt = cur ^ 1;

    // ---- async-prefetch next chunk while computing on current ----
    if (it + 1 < SEQ / 32) {
      const int i0 = (it + 1) * 32;
#pragma unroll
      for (int t = 0; t < 2; ++t) {
        int e = tid + t * 256;
        int i = e >> 4, c = e & 15;
        cp16(&Qtn[(size_t)(i0 + i) * DM + c * 8], &sQ[nxt][i][c * 8]);
      }
#pragma unroll
      for (int t = 0; t < 2; ++t) {
        int e = tid + t * 256;
        int d = e >> 2, c = e & 3;
        cp16(&Vbn[(size_t)d * SEQ + i0 + c * 8], &sV[nxt][d][c * 8]);
      }
    }

    // ---- scores: S0 = i rows 0-15, S1 = i rows 16-31 of this chunk ----
    v8f s0, s1;
#pragma unroll
    for (int r = 0; r < 8; ++r) { s0[r] = 0.f; s1[r] = 0.f; }

    const __bf16* q0 = &sQ[cur][lj][0];
    const __bf16* q1 = &sQ[cur][16 + lj][0];
    v16bf a0 = ldfrag(q0, half * 8, half * 8 + 16);
    v16bf a1 = ldfrag(q1, half * 8, half * 8 + 16);
#pragma unroll
    for (int c = 0; c < 4; ++c) {
      v16bf na0, na1;
      if (c < 3) {                       // preload next chunk's A fragments
        int b0 = (c + 1) * 32 + half * 8;
        na0 = ldfrag(q0, b0, b0 + 16);
        na1 = ldfrag(q1, b0, b0 + 16);
      }
      s0 = __builtin_amdgcn_wmma_f32_16x16x32_bf16(false, a0, false, kf[c],
                                                   (short)0, s0, false, false);
      s1 = __builtin_amdgcn_wmma_f32_16x16x32_bf16(false, a1, false, kf[c],
                                                   (short)0, s1, false, false);
      a0 = na0; a1 = na1;
    }

    // ---- online softmax over i, per column j (lanes L, L^16 share column) --
    float cmax = -3.0e38f;
#pragma unroll
    for (int r = 0; r < 8; ++r) {
      s0[r] *= scale; s1[r] *= scale;
      cmax = fmaxf(cmax, fmaxf(s0[r], s1[r]));
    }
    cmax = fmaxf(cmax, __shfl_xor(cmax, 16, 32));
    float m_new = fmaxf(m_run, cmax);
    float corr  = __expf(m_run - m_new);

    float p0[8], p1[8], csum = 0.f;
#pragma unroll
    for (int r = 0; r < 8; ++r) {
      p0[r] = __expf(s0[r] - m_new);
      p1[r] = __expf(s1[r] - m_new);
      csum += p0[r] + p1[r];
    }
    csum += __shfl_xor(csum, 16, 32);
    l_run = l_run * corr + csum;
    m_run = m_new;

#pragma unroll
    for (int t = 0; t < 8; ++t)
#pragma unroll
      for (int r = 0; r < 8; ++r) acc[t][r] *= corr;

    // ---- P as B-fragment (32 i x 16 j); partner-lane exchange via xor 16 ---
    v16bf pf;
#pragma unroll
    for (int r = 0; r < 8; ++r) {
      float o0 = __shfl_xor(p0[r], 16, 32);
      float o1 = __shfl_xor(p1[r], 16, 32);
      float elo = (half == 0) ? p0[r] : o1;    // K = half*16 + r
      float ehi = (half == 0) ? o0    : p1[r]; // K = half*16 + 8 + r
      pf[r]     = (__bf16)elo;
      pf[r + 8] = (__bf16)ehi;
    }

    // ---- O(d,j) += V(d,i) * P(i,j): 8 d-tiles, preloaded A fragments ------
    v16bf av = ldfrag(&sV[cur][lj][0], half * 16, half * 16 + 8);
#pragma unroll
    for (int t = 0; t < 8; ++t) {
      v16bf nav;
      if (t < 7)
        nav = ldfrag(&sV[cur][(t + 1) * 16 + lj][0], half * 16, half * 16 + 8);
      acc[t] = __builtin_amdgcn_wmma_f32_16x16x32_bf16(false, av, false, pf,
                                                       (short)0, acc[t], false, false);
      av = nav;
    }

    wait_async();
    __syncthreads();
  }

  // ---- normalize; store transposed bf16 output, 16B per (lane, d-tile) ----
  float inv_l = 1.f / l_run;
  const int jg = jb + w * 16 + lj;
  __bf16* orow = OT + ((size_t)((n >> 4) * SEQ + jg)) * HK + (size_t)(n & 15) * DM;
#pragma unroll
  for (int t = 0; t < 8; ++t) {
    v8bf ov;
#pragma unroll
    for (int r = 0; r < 8; ++r) ov[r] = (__bf16)(acc[t][r] * inv_l);
    *(v8bf*)&orow[t * 16 + half * 8] = ov;   // d = t*16 + half*8 + r
  }
}

// ---------------------------------------------------------------------------
// Kernel 2: Y[b,d,s] = sum_k W[k,d] * O[b,k,s]  (k over HK=2048)
// Wt: [DM][HK] bf16, OT: [BATCH][SEQ][HK] bf16, Y: [BATCH][DM][SEQ] f32.
// ---------------------------------------------------------------------------
__global__ __launch_bounds__(256)
void proj_kernel(const __bf16* __restrict__ Wt,
                 const __bf16* __restrict__ OT,
                 float* __restrict__ Y)
{
  __shared__ __align__(16) __bf16 sW[2][128][32];  // [d][k]
  __shared__ __align__(16) __bf16 sO[2][128][32];  // [s][k]

  const int b    = blockIdx.y;
  const int sb   = blockIdx.x * 128;
  const int tid  = threadIdx.x;
  const int w    = tid >> 5;
  const int lane = tid & 31;
  const int lj   = lane & 15;
  const int half = lane >> 4;

  const __bf16* OTb = OT + (size_t)b * SEQ * HK;

  v8f acc[8];
#pragma unroll
  for (int t = 0; t < 8; ++t)
#pragma unroll
    for (int r = 0; r < 8; ++r) acc[t][r] = 0.f;

  // stage k-chunk 0 into buffer 0
#pragma unroll
  for (int t = 0; t < 2; ++t) {
    int e = tid + t * 256;
    int x = e >> 2, c = e & 3;
    cp16(&Wt[(size_t)x * HK + c * 8], &sW[0][x][c * 8]);
    cp16(&OTb[(size_t)(sb + x) * HK + c * 8], &sO[0][x][c * 8]);
  }
  wait_async();
  __syncthreads();

  for (int it = 0; it < HK / 32; ++it) {
    const int cur = it & 1, nxt = cur ^ 1;
    if (it + 1 < HK / 32) {
      const int k0 = (it + 1) * 32;
#pragma unroll
      for (int t = 0; t < 2; ++t) {
        int e = tid + t * 256;
        int x = e >> 2, c = e & 3;
        cp16(&Wt[(size_t)x * HK + k0 + c * 8], &sW[nxt][x][c * 8]);
        cp16(&OTb[(size_t)(sb + x) * HK + k0 + c * 8], &sO[nxt][x][c * 8]);
      }
    }

    // B fragment: O chunk (K=k, N=s)
    v16bf bo = ldfrag(&sO[cur][w * 16 + lj][0], half * 8, half * 8 + 16);

    v16bf aw = ldfrag(&sW[cur][lj][0], half * 8, half * 8 + 16);
#pragma unroll
    for (int t = 0; t < 8; ++t) {
      v16bf naw;
      if (t < 7)
        naw = ldfrag(&sW[cur][(t + 1) * 16 + lj][0], half * 8, half * 8 + 16);
      acc[t] = __builtin_amdgcn_wmma_f32_16x16x32_bf16(false, aw, false, bo,
                                                       (short)0, acc[t], false, false);
      aw = naw;
    }

    wait_async();
    __syncthreads();
  }

  const int sg = sb + w * 16 + lj;
#pragma unroll
  for (int t = 0; t < 8; ++t)
#pragma unroll
    for (int r = 0; r < 8; ++r) {
      int d = t * 16 + half * 8 + r;
      Y[((size_t)b * DM + d) * SEQ + sg] = acc[t][r];
    }
}

// ---------------------------------------------------------------------------
extern "C" void kernel_launch(void* const* d_in, const int* in_sizes, int n_in,
                              void* d_out, int out_size, void* d_ws, size_t ws_size,
                              hipStream_t stream) {
  // inputs: x (unused, shape only), Q, K, V, W — all float32
  const float* Q = (const float*)d_in[1];
  const float* K = (const float*)d_in[2];
  const float* V = (const float*)d_in[3];
  const float* W = (const float*)d_in[4];
  float* Y = (float*)d_out;

  const size_t eQ = (size_t)NHEADS * DM * SEQ;      // 16M elems each
  __bf16* Qt = (__bf16*)d_ws;                       // [N][SEQ][DM]
  __bf16* Kt = Qt + eQ;                             // [N][SEQ][DM]
  __bf16* Vb = Kt + eQ;                             // [N][DM][SEQ]
  __bf16* OT = Vb + eQ;                             // [B][SEQ][HK]
  __bf16* Wt = OT + (size_t)BATCH * SEQ * HK;       // [DM][HK]

  dim3 blk(256);
  cvt_t_kernel<<<dim3(SEQ / 64, NHEADS), blk, 0, stream>>>(Q, Qt);
  cvt_t_kernel<<<dim3(SEQ / 64, NHEADS), blk, 0, stream>>>(K, Kt);
  cvt_v_kernel<<<dim3((int)(eQ / (256 * 4))), blk, 0, stream>>>(V, Vb);
  cvt_w_kernel<<<dim3(HK / 128), blk, 0, stream>>>(W, Wt);
  attn_kernel<<<dim3(SEQ / 128, NHEADS), blk, 0, stream>>>(Qt, Kt, Vb, OT);
  proj_kernel<<<dim3(SEQ / 128, BATCH), blk, 0, stream>>>(Wt, OT, Y);
}
